// SpatialProcessor_65000035057948
// MI455X (gfx1250) — matrix operand order repaired and checked by reading, not compiled, not run on hardware
//
#include <hip/hip_runtime.h>
#include <hip/hip_bf16.h>

// ---------------------------------------------------------------------------
// SpatialProcessor (2-layer graph attention) for MI455X / gfx1250.
//
// Mask is provably a no-op (adj = sigmoid(cos) in (0.268, 0.732] U {1}, never
// 0), so this is dense flash-attention. All GEMMs use v_wmma_f32_16x16x32_bf16
// with f32 accumulation; softmax/residual/LayerNorm in fp32. K/V tiles are
// staged into LDS with double-buffered GLOBAL_LOAD_ASYNC_TO_LDS_B128
// (ASYNCcnt), shared by the 4 waves of each block.
// ---------------------------------------------------------------------------

#define B_SZ   4
#define N_CTX  2048
#define FIN    64
#define DTOT   128

typedef __attribute__((ext_vector_type(16))) __bf16 bf16x16;
typedef __attribute__((ext_vector_type(8)))  float  f32x8;

static __device__ __forceinline__ __bf16 f2bf(float f) {
  unsigned u = __float_as_uint(f);
  unsigned r = (u + 0x7FFFu + ((u >> 16) & 1u)) >> 16;   // round-to-nearest-even
  union { unsigned short s; __bf16 b; } cv;
  cv.s = (unsigned short)r;
  return cv.b;
}

union AB32 { uint4 q[2]; bf16x16 v; };

// A-frag (16x32, 16-bit): lane l -> row l&15; 16 contiguous K at offset 16*part
static __device__ __forceinline__ bf16x16 load_afrag(const __bf16* rowbase, int part) {
  AB32 u;
  u.q[0] = *(const uint4*)(rowbase + 16 * part);
  u.q[1] = *(const uint4*)(rowbase + 16 * part + 8);
  return u.v;
}
// B-frag (32x16, 16-bit): lane l -> col l&15; 16 contiguous K at offset 16*part
static __device__ __forceinline__ bf16x16 load_bfrag(const __bf16* p) {
  AB32 u;
  u.q[0] = *(const uint4*)(p);
  u.q[1] = *(const uint4*)(p + 8);
  return u.v;
}

static __device__ __forceinline__ f32x8 wmma_bf16(bf16x16 a, bf16x16 b, f32x8 c) {
  return __builtin_amdgcn_wmma_f32_16x16x32_bf16(false, a, false, b, (short)0, c,
                                                 false, false);
}

// Async global->LDS copy (16B per lane), tracked by ASYNCcnt.
static __device__ __forceinline__ void async_copy_b128(uint32_t lds_off,
                                                       uint64_t gaddr) {
  asm volatile("global_load_async_to_lds_b128 %0, %1, off"
               :: "v"(lds_off), "v"(gaddr)
               : "memory");
}
static __device__ __forceinline__ void wait_async0() {
  asm volatile("s_wait_asynccnt 0" ::: "memory");
}
static __device__ __forceinline__ uint32_t lds_addr32(const void* p) {
  return (uint32_t)(uintptr_t)p;   // low 32 bits of a generic shared ptr = LDS offset
}

// 16-lane reductions inside each half of the wave32 (C-layout row reductions)
static __device__ __forceinline__ float rmax16(float v) {
#pragma unroll
  for (int m = 1; m < 16; m <<= 1) v = fmaxf(v, __shfl_xor(v, m, 32));
  return v;
}
static __device__ __forceinline__ float rsum16(float v) {
#pragma unroll
  for (int m = 1; m < 16; m <<= 1) v += __shfl_xor(v, m, 32);
  return v;
}

// ---------------------------------------------------------------------------
// prep kernels
// ---------------------------------------------------------------------------
__global__ void cvt_f32_bf16(const float* __restrict__ src,
                             __bf16* __restrict__ dst, int n) {
  int i = blockIdx.x * blockDim.x + threadIdx.x;
  if (i < n) dst[i] = f2bf(src[i]);
}

// W (K, Nc) fp32 row-major  ->  WT (Nc, K) bf16 row-major (for B-frags)
__global__ void transpose_w(const float* __restrict__ src,
                            __bf16* __restrict__ dst, int K, int Nc) {
  int i = blockIdx.x * blockDim.x + threadIdx.x;
  if (i < K * Nc) {
    int k = i / Nc, n = i - k * Nc;
    dst[(size_t)n * K + k] = f2bf(src[i]);
  }
}

// ---------------------------------------------------------------------------
// Projection GEMM: (rows x K) bf16 @ (K x 128) -> one 16x16 tile per wave.
// ---------------------------------------------------------------------------
template <int K>
__global__ void proj_kernel(const __bf16* __restrict__ X,
                            const __bf16* __restrict__ WT,
                            const float* __restrict__ bias,
                            __bf16* __restrict__ outBF,
                            float* __restrict__ outF32,
                            __bf16* __restrict__ outVT) {
  int wave = blockIdx.x * (blockDim.x >> 5) + (threadIdx.x >> 5);
  int lane = threadIdx.x & 31;
  int ct = wave & 7;        // 8 column tiles of 16 (out width 128)
  int rt = wave >> 3;       // row tile
  int m = lane & 15, part = lane >> 4;

  const __bf16* arow = X + (size_t)(rt * 16 + m) * K;
  const __bf16* brow = WT + (size_t)(ct * 16 + m) * K;
  f32x8 acc = {};
#pragma unroll
  for (int k0 = 0; k0 < K; k0 += 32) {
    bf16x16 a = load_afrag(arow + k0, part);
    bf16x16 b = load_bfrag(brow + k0 + 16 * part);
    acc = wmma_bf16(a, b, acc);
  }

  int col = ct * 16 + m;
  float bv = bias ? bias[col] : 0.0f;
#pragma unroll
  for (int i = 0; i < 8; ++i) {
    int row = rt * 16 + i + 8 * part;   // C layout: row = i + 8*(lane>>4)
    float v = acc[i] + bv;
    if (outBF)  outBF[(size_t)row * DTOT + col] = f2bf(v);
    if (outF32) outF32[(size_t)row * DTOT + col] = v;
    if (outVT) {                        // (b, col, n) : K-contiguous for PV B-frags
      int b_ = row >> 11;               // row / 2048
      int n  = row & 2047;
      outVT[((size_t)(b_ * DTOT + col)) * N_CTX + n] = f2bf(v);
    }
  }
}

// ---------------------------------------------------------------------------
// Flash attention. One wave owns a 16-row tile; the 4 waves of a block share
// one (b, h) K/V stream staged into LDS by double-buffered async copies.
//   Q, Km: (B*N, 128) bf16 (head h = columns h*D .. h*D+D-1)
//   VT:    (B, 128, N) bf16
// FINAL=false : ReLU, bf16 out (layer-1 hidden)
// FINAL=true  : + residual Pproj, LayerNorm(eps=1e-3), fp32 out
// ---------------------------------------------------------------------------
template <int D, int H, bool FINAL>
__global__ void attn_kernel(const __bf16* __restrict__ Q,
                            const __bf16* __restrict__ Km,
                            const __bf16* __restrict__ VT,
                            const float* __restrict__ Pproj,
                            const float* __restrict__ ln_g,
                            const float* __restrict__ ln_b,
                            __bf16* __restrict__ outBF,
                            float* __restrict__ outF32) {
  constexpr int NT = D / 16;            // output column tiles
  constexpr int KC = D / 32;            // K-depth chunks for the score GEMM
  constexpr int SPR = D / 8;            // 16B segments per K-tile row
  const float scale = (D == 32) ? 0.17677669529663687f : 0.08838834764831843f;

  __shared__ __align__(16) __bf16 ldsK[2][32 * D];   // K tile: 32 rows x D
  __shared__ __align__(16) __bf16 ldsV[2][D * 32];   // V tile: D rows x 32 j
  __shared__ __align__(16) __bf16 ldsP[4][16][32];   // per-wave P staging

  int wv = threadIdx.x >> 5;
  int lane = threadIdx.x & 31;
  int wave = blockIdx.x * 4 + wv;
  int rt = wave & 127;                  // N/16 = 128 row tiles
  int h = (wave >> 7) % H;
  int b = wave / (128 * H);
  int m = lane & 15, part = lane >> 4;
  int row0 = rt * 16;
  int t = wv * 32 + lane;               // 0..127: block-wide lane id for copies

  // Loop-invariant Q fragments
  bf16x16 aq[KC];
  {
    const __bf16* qrow = Q + ((size_t)(b * N_CTX + row0 + m)) * DTOT + h * D;
#pragma unroll
    for (int kc = 0; kc < KC; ++kc) aq[kc] = load_afrag(qrow + kc * 32, part);
  }

  // Per-lane async-copy source addresses (advance by constants per chunk)
  // K tile chunk c (c = t + 128*r): row = c/SPR, seg = c%SPR
  // V tile chunk c:                 d   = c>>2,  seg = c&3
  uint64_t gK[KC], gV[KC];
  uint32_t lK[KC], lV[KC];
#pragma unroll
  for (int r = 0; r < KC; ++r) {
    int c = t + 128 * r;
    int krow = c / SPR, kseg = c % SPR;
    gK[r] = (uint64_t)(uintptr_t)(Km + ((size_t)(b * N_CTX + krow)) * DTOT + h * D) +
            (uint64_t)kseg * 16;
    lK[r] = (uint32_t)(krow * (D * 2) + kseg * 16);
    int d = c >> 2, vseg = c & 3;
    gV[r] = (uint64_t)(uintptr_t)(VT + ((size_t)(b * DTOT + h * D + d)) * N_CTX) +
            (uint64_t)vseg * 16;
    lV[r] = (uint32_t)(d * 64 + vseg * 16);
  }
  uint32_t kbase[2] = { lds_addr32(&ldsK[0][0]), lds_addr32(&ldsK[1][0]) };
  uint32_t vbase[2] = { lds_addr32(&ldsV[0][0]), lds_addr32(&ldsV[1][0]) };

  auto issue = [&](int j0, int buf) {
#pragma unroll
    for (int r = 0; r < KC; ++r) {
      async_copy_b128(kbase[buf] + lK[r], gK[r] + (uint64_t)j0 * (DTOT * 2));
      async_copy_b128(vbase[buf] + lV[r], gV[r] + (uint64_t)j0 * 2);
    }
  };

  float mrun[8], lsum[8];
  f32x8 o[NT];
  f32x8 zero = {};
#pragma unroll
  for (int i = 0; i < 8; ++i) { mrun[i] = -3.0e38f; lsum[i] = 0.0f; }
#pragma unroll
  for (int nt = 0; nt < NT; ++nt) o[nt] = zero;

  issue(0, 0);                          // prefetch first chunk

  for (int j0 = 0; j0 < N_CTX; j0 += 32) {
    int cur = (j0 >> 5) & 1;
    wait_async0();                      // this wave's quarter of buffer `cur`
    __syncthreads();                    // all quarters visible to all waves
    if (j0 + 32 < N_CTX) issue(j0 + 32, cur ^ 1);

    // ---- scores: two 16x16 tiles (cols j0..+15, j0+16..+31) from LDS K tile
    f32x8 s0 = zero, s1 = zero;
    const __bf16* lk = &ldsK[cur][0];
#pragma unroll
    for (int kc = 0; kc < KC; ++kc)
      s0 = wmma_bf16(aq[kc], load_bfrag(lk + m * D + kc * 32 + 16 * part), s0);
#pragma unroll
    for (int kc = 0; kc < KC; ++kc)
      s1 = wmma_bf16(aq[kc], load_bfrag(lk + (16 + m) * D + kc * 32 + 16 * part), s1);

    // ---- online softmax (row reductions = 16-lane half-wave shuffles)
    float p0[8], p1[8], cf[8];
#pragma unroll
    for (int i = 0; i < 8; ++i) {
      float a0 = s0[i] * scale, a1 = s1[i] * scale;
      float mt = rmax16(fmaxf(a0, a1));
      float mn = fmaxf(mrun[i], mt);
      cf[i] = __expf(mrun[i] - mn);
      mrun[i] = mn;
      p0[i] = __expf(a0 - mn);
      p1[i] = __expf(a1 - mn);
      lsum[i] = lsum[i] * cf[i] + rsum16(p0[i] + p1[i]);
    }
#pragma unroll
    for (int nt = 0; nt < NT; ++nt)
#pragma unroll
      for (int i = 0; i < 8; ++i) o[nt][i] *= cf[i];

    // ---- C-layout -> A-layout for P via LDS (in-order per wave; fence dscnt)
#pragma unroll
    for (int i = 0; i < 8; ++i) {
      int r = i + 8 * part;
      ldsP[wv][r][m]      = f2bf(p0[i]);
      ldsP[wv][r][16 + m] = f2bf(p1[i]);
    }
    asm volatile("s_wait_dscnt 0" ::: "memory");
    bf16x16 pf = load_afrag(&ldsP[wv][m][0], part);

    // ---- O += P(16x32) @ V(32x16) per column tile from LDS V tile
#pragma unroll
    for (int nt = 0; nt < NT; ++nt)
      o[nt] = wmma_bf16(pf, load_bfrag(&ldsV[cur][(nt * 16 + m) * 32 + 16 * part]),
                        o[nt]);

    __syncthreads();                    // buffer `cur` free to overwrite
  }

  float inv[8];
#pragma unroll
  for (int i = 0; i < 8; ++i) inv[i] = 1.0f / lsum[i];

  if constexpr (!FINAL) {
#pragma unroll
    for (int nt = 0; nt < NT; ++nt) {
      int col = h * D + nt * 16 + m;
#pragma unroll
      for (int i = 0; i < 8; ++i) {
        int row = row0 + i + 8 * part;
        float v = fmaxf(o[nt][i] * inv[i], 0.0f);   // ReLU
        outBF[((size_t)(b * N_CTX + row)) * DTOT + col] = f2bf(v);
      }
    }
  } else {
    // residual + LayerNorm over the 128 features of each row
    float y[NT][8];
#pragma unroll
    for (int nt = 0; nt < NT; ++nt) {
      int col = nt * 16 + m;
#pragma unroll
      for (int i = 0; i < 8; ++i) {
        int row = row0 + i + 8 * part;
        y[nt][i] = o[nt][i] * inv[i] +
                   Pproj[((size_t)(b * N_CTX + row)) * DTOT + col];
      }
    }
    float mu[8], rstd[8];
#pragma unroll
    for (int i = 0; i < 8; ++i) {
      float s = 0.0f;
#pragma unroll
      for (int nt = 0; nt < NT; ++nt) s += y[nt][i];
      mu[i] = rsum16(s) * (1.0f / 128.0f);
      float q = 0.0f;
#pragma unroll
      for (int nt = 0; nt < NT; ++nt) { float d = y[nt][i] - mu[i]; q += d * d; }
      rstd[i] = rsqrtf(rsum16(q) * (1.0f / 128.0f) + 1e-3f);
    }
#pragma unroll
    for (int nt = 0; nt < NT; ++nt) {
      int col = nt * 16 + m;
      float g = ln_g[col], be = ln_b[col];
#pragma unroll
      for (int i = 0; i < 8; ++i) {
        int row = row0 + i + 8 * part;
        outF32[((size_t)(b * N_CTX + row)) * DTOT + col] =
            (y[nt][i] - mu[i]) * rstd[i] * g + be;
      }
    }
  }
}

// ---------------------------------------------------------------------------
extern "C" void kernel_launch(void* const* d_in, const int* in_sizes, int n_in,
                              void* d_out, int out_size, void* d_ws, size_t ws_size,
                              hipStream_t stream) {
  const float* x      = (const float*)d_in[0];
  // d_in[1] = emb is intentionally unused: adj = sigmoid(cos_sim) is never 0,
  // so the (adj == 0 -> -1e9) mask in the reference is a no-op.
  const float* q1_w   = (const float*)d_in[2];
  const float* q1_b   = (const float*)d_in[3];
  const float* k1_w   = (const float*)d_in[4];
  const float* k1_b   = (const float*)d_in[5];
  const float* v1_w   = (const float*)d_in[6];
  const float* v1_b   = (const float*)d_in[7];
  const float* q2_w   = (const float*)d_in[8];
  const float* q2_b   = (const float*)d_in[9];
  const float* k2_w   = (const float*)d_in[10];
  const float* k2_b   = (const float*)d_in[11];
  const float* v2_w   = (const float*)d_in[12];
  const float* v2_b   = (const float*)d_in[13];
  const float* proj_w = (const float*)d_in[14];
  const float* ln_g   = (const float*)d_in[15];
  const float* ln_b   = (const float*)d_in[16];

  const int ROWS = B_SZ * N_CTX;        // 8192

  char* ws = (char*)d_ws;
  size_t off = 0;
  auto alloc = [&](size_t bytes) -> void* {
    void* p = ws + off;
    off = (off + bytes + 255) & ~(size_t)255;
    return p;
  };

  __bf16* xb    = (__bf16*)alloc((size_t)ROWS * FIN * 2);
  __bf16* wtq1  = (__bf16*)alloc((size_t)DTOT * FIN * 2);
  __bf16* wtk1  = (__bf16*)alloc((size_t)DTOT * FIN * 2);
  __bf16* wtv1  = (__bf16*)alloc((size_t)DTOT * FIN * 2);
  __bf16* wtp   = (__bf16*)alloc((size_t)DTOT * FIN * 2);
  __bf16* wtq2  = (__bf16*)alloc((size_t)DTOT * DTOT * 2);
  __bf16* wtk2  = (__bf16*)alloc((size_t)DTOT * DTOT * 2);
  __bf16* wtv2  = (__bf16*)alloc((size_t)DTOT * DTOT * 2);
  __bf16* Q1    = (__bf16*)alloc((size_t)ROWS * DTOT * 2);
  __bf16* K1    = (__bf16*)alloc((size_t)ROWS * DTOT * 2);
  __bf16* VT1   = (__bf16*)alloc((size_t)B_SZ * DTOT * N_CTX * 2);
  float*  Pp    = (float*) alloc((size_t)ROWS * DTOT * 4);
  __bf16* h1    = (__bf16*)alloc((size_t)ROWS * DTOT * 2);
  __bf16* Q2    = (__bf16*)alloc((size_t)ROWS * DTOT * 2);
  __bf16* K2    = (__bf16*)alloc((size_t)ROWS * DTOT * 2);
  __bf16* VT2   = (__bf16*)alloc((size_t)B_SZ * DTOT * N_CTX * 2);

  // --- prep: fp32 -> bf16 input, transposed bf16 weights (for B-frag layout)
  {
    int n = ROWS * FIN;
    cvt_f32_bf16<<<(n + 255) / 256, 256, 0, stream>>>(x, xb, n);
    int nw1 = FIN * DTOT, nw2 = DTOT * DTOT;
    transpose_w<<<(nw1 + 255) / 256, 256, 0, stream>>>(q1_w,   wtq1, FIN, DTOT);
    transpose_w<<<(nw1 + 255) / 256, 256, 0, stream>>>(k1_w,   wtk1, FIN, DTOT);
    transpose_w<<<(nw1 + 255) / 256, 256, 0, stream>>>(v1_w,   wtv1, FIN, DTOT);
    transpose_w<<<(nw1 + 255) / 256, 256, 0, stream>>>(proj_w, wtp,  FIN, DTOT);
    transpose_w<<<(nw2 + 255) / 256, 256, 0, stream>>>(q2_w,   wtq2, DTOT, DTOT);
    transpose_w<<<(nw2 + 255) / 256, 256, 0, stream>>>(k2_w,   wtk2, DTOT, DTOT);
    transpose_w<<<(nw2 + 255) / 256, 256, 0, stream>>>(v2_w,   wtv2, DTOT, DTOT);
  }

  // --- layer-1 projections + residual projection (fp32)
  const int PROJ_BLOCKS = (ROWS / 16) * 8 / 4;  // 4 waves per 128-thread block
  proj_kernel<FIN><<<PROJ_BLOCKS, 128, 0, stream>>>(xb, wtq1, q1_b, Q1, nullptr, nullptr);
  proj_kernel<FIN><<<PROJ_BLOCKS, 128, 0, stream>>>(xb, wtk1, k1_b, K1, nullptr, nullptr);
  proj_kernel<FIN><<<PROJ_BLOCKS, 128, 0, stream>>>(xb, wtv1, v1_b, nullptr, nullptr, VT1);
  proj_kernel<FIN><<<PROJ_BLOCKS, 128, 0, stream>>>(xb, wtp, nullptr, nullptr, Pp, nullptr);

  // --- layer-1 flash attention (4 heads, d=32) + ReLU -> bf16 hidden
  attn_kernel<32, 4, false><<<(B_SZ * 4 * (N_CTX / 16)) / 4, 128, 0, stream>>>(
      Q1, K1, VT1, nullptr, nullptr, nullptr, h1, nullptr);

  // --- layer-2 projections (K = 128)
  proj_kernel<DTOT><<<PROJ_BLOCKS, 128, 0, stream>>>(h1, wtq2, q2_b, Q2, nullptr, nullptr);
  proj_kernel<DTOT><<<PROJ_BLOCKS, 128, 0, stream>>>(h1, wtk2, k2_b, K2, nullptr, nullptr);
  proj_kernel<DTOT><<<PROJ_BLOCKS, 128, 0, stream>>>(h1, wtv2, v2_b, nullptr, nullptr, VT2);

  // --- layer-2 flash attention (1 head, d=128) + residual + LayerNorm -> out
  attn_kernel<128, 1, true><<<(B_SZ * 1 * (N_CTX / 16)) / 4, 128, 0, stream>>>(
      Q2, K2, VT2, Pp, ln_g, ln_b, nullptr, (float*)d_out);
}
